// PureDynamicTransformer_22445499089544
// MI455X (gfx1250) — matrix operand order repaired and checked by reading, not compiled, not run on hardware
//
#include <hip/hip_runtime.h>
#include <math.h>

#define VOCAB    32000
#define DIM      1024
#define HEADS    16
#define LAYERS   6
#define SEQ      2048
#define BATCH    2
#define HEAD_DIM 64
#define DFF      4096
#define NTOK     (BATCH * SEQ)   // 4096 token rows

typedef __attribute__((ext_vector_type(16))) __bf16 v16bf;
typedef __attribute__((ext_vector_type(2)))  __bf16 v2bf;
typedef __attribute__((ext_vector_type(8)))  float  v8f;
typedef __attribute__((ext_vector_type(2)))  float  v2f;

// fp32 -> bf16 via native conversion
static __device__ __forceinline__ __bf16 f2bf(float f) {
    return (__bf16)f;
}
// two fp32 -> packed bf16 dword; vector fptrunc selects one v_cvt_pk_bf16_f32
static __device__ __forceinline__ unsigned pack2(float lo, float hi) {
    v2f x; x[0] = lo; x[1] = hi;
    v2bf t = __builtin_convertvector(x, v2bf);
    return __builtin_bit_cast(unsigned, t);
}

// A-matrix (16x32 bf16) element->K map per ISA 7.12.2:
// VGPR i (pair): K = 2i (+8 if i>=4) + half*8, +j within pair
static __device__ __forceinline__ int kA(int e, int half) {
    int i = e >> 1;
    return 2 * i + (i < 4 ? 0 : 8) + half * 8 + (e & 1);
}

static __device__ __forceinline__ v8f wmma_bf16(v16bf a, v16bf b, v8f c) {
    return __builtin_amdgcn_wmma_f32_16x16x32_bf16(false, a, false, b,
                                                   (short)0, c, false, false);
}

static __device__ __forceinline__ float gelu_exact(float x) {
    return 0.5f * x * (1.0f + erff(x * 0.70710678118654752f));
}

// ---------------------------------------------------------------- embedding
__global__ void k_embed(const int* __restrict__ ids,
                        const float* __restrict__ emb,
                        const float* __restrict__ pos,
                        float* __restrict__ X) {
    int tok = blockIdx.x;
    int s   = tok % SEQ;
    int id  = ids[tok];
    const float* e = emb + (size_t)id * DIM;
    const float* p = pos + (size_t)s * DIM;
    float* x = X + (size_t)tok * DIM;
    for (int d = threadIdx.x; d < DIM; d += blockDim.x)
        x[d] = e[d] + p[d];
}

// ---------------------------------------------------------------- layernorm
__global__ __launch_bounds__(256) void k_ln(const float* __restrict__ in,
                                            const float* __restrict__ sc,
                                            const float* __restrict__ bi,
                                            float* __restrict__ out) {
    __shared__ float red[256];
    int row = blockIdx.x;
    const float* x = in + (size_t)row * DIM;
    float s = 0.f, s2 = 0.f;
    for (int d = threadIdx.x; d < DIM; d += 256) {
        float v = x[d]; s += v; s2 += v * v;
    }
    red[threadIdx.x] = s; __syncthreads();
    for (int o = 128; o > 0; o >>= 1) {
        if (threadIdx.x < o) red[threadIdx.x] += red[threadIdx.x + o];
        __syncthreads();
    }
    float mean = red[0] * (1.0f / DIM);
    __syncthreads();
    red[threadIdx.x] = s2; __syncthreads();
    for (int o = 128; o > 0; o >>= 1) {
        if (threadIdx.x < o) red[threadIdx.x] += red[threadIdx.x + o];
        __syncthreads();
    }
    float var  = red[0] * (1.0f / DIM) - mean * mean;
    float rstd = rsqrtf(var + 1e-5f);
    float* y = out + (size_t)row * DIM;
    for (int d = threadIdx.x; d < DIM; d += 256)
        y[d] = (x[d] - mean) * rstd * sc[d] + bi[d];
}

// ------------------------------------------------- fused GEMM: C = act(A*W^T + b) (+res)
// A:[M,K] row-major, W:[N,K] row-major. M mult of 128, N mult of 64, K mult of 32.
// 256 threads = 8 waves; block tile 128x64; wave tile 32x32 (4 WMMA accumulators).
// Software pipelined: next K-tile's global loads issued before this tile's WMMAs.
__global__ __launch_bounds__(256) void k_gemm(const float* __restrict__ A,
                                              const float* __restrict__ W,
                                              const float* __restrict__ bias,
                                              const float* __restrict__ residual,
                                              float* __restrict__ C,
                                              int M, int N, int K, int act_gelu) {
    __shared__ __align__(16) __bf16 As[128][32];
    __shared__ __align__(16) __bf16 Bs[64][32];
    int tid  = threadIdx.x;
    int m0   = blockIdx.y * 128, n0 = blockIdx.x * 64;
    int lane = tid & 31, half = lane >> 4, r = lane & 15;
    int w    = tid >> 5;
    int wm   = (w & 3) * 32;     // 4 waves along M, 32 rows each
    int wn   = (w >> 2) * 32;    // 2 waves along N, 32 cols each

    // staging registers: A 4 x float4, B 2 x float4 per thread
    float4 ra[4], rb[2];
    auto gload = [&](int k0) {
        for (int j = 0; j < 4; ++j) {
            int f = tid + 256 * j;              // float4 index: 8 per A row
            int row = f >> 3, coff = (f & 7) * 4;
            ra[j] = *(const float4*)(A + (size_t)(m0 + row) * K + k0 + coff);
        }
        for (int j = 0; j < 2; ++j) {
            int f = tid + 256 * j;
            int row = f >> 3, coff = (f & 7) * 4;
            rb[j] = *(const float4*)(W + (size_t)(n0 + row) * K + k0 + coff);
        }
    };
    auto lstore = [&]() {
        for (int j = 0; j < 4; ++j) {
            int f = tid + 256 * j;
            int row = f >> 3, coff = (f & 7) * 4;
            uint2 p; p.x = pack2(ra[j].x, ra[j].y); p.y = pack2(ra[j].z, ra[j].w);
            *(uint2*)&As[row][coff] = p;
        }
        for (int j = 0; j < 2; ++j) {
            int f = tid + 256 * j;
            int row = f >> 3, coff = (f & 7) * 4;
            uint2 p; p.x = pack2(rb[j].x, rb[j].y); p.y = pack2(rb[j].z, rb[j].w);
            *(uint2*)&Bs[row][coff] = p;
        }
    };

    v8f acc[2][2];
    acc[0][0] = v8f{}; acc[0][1] = v8f{}; acc[1][0] = v8f{}; acc[1][1] = v8f{};

    gload(0);
    for (int k0 = 0; k0 < K; k0 += 32) {
        lstore();
        __syncthreads();
        if (k0 + 32 < K) gload(k0 + 32);       // in flight during WMMAs below
        v16bf a0, a1, b0, b1;
        for (int e = 0; e < 16; ++e) {
            int ka = kA(e, half), kb = kA(e, half);
            a0[e] = As[wm + r][ka];
            a1[e] = As[wm + 16 + r][ka];
            b0[e] = Bs[wn + r][kb];
            b1[e] = Bs[wn + 16 + r][kb];
        }
        acc[0][0] = wmma_bf16(a0, b0, acc[0][0]);
        acc[0][1] = wmma_bf16(a0, b1, acc[0][1]);
        acc[1][0] = wmma_bf16(a1, b0, acc[1][0]);
        acc[1][1] = wmma_bf16(a1, b1, acc[1][1]);
        __syncthreads();
    }
    for (int mi = 0; mi < 2; ++mi)
        for (int ni = 0; ni < 2; ++ni)
            for (int v = 0; v < 8; ++v) {
                int row = m0 + wm + mi * 16 + v + 8 * half;  // C: M = vgpr + 8*half
                int col = n0 + wn + ni * 16 + r;             //    N = lane&15
                float x = acc[mi][ni][v];
                if (bias)     x += bias[col];
                if (act_gelu) x = gelu_exact(x);
                if (residual) x += residual[(size_t)row * N + col];
                C[(size_t)row * N + col] = x;
            }
}

// ------------------------------------------------- flash attention
// 128 threads = 4 waves per block; block covers 64 query rows (16 per wave);
// K/V chunks (32 kv rows) staged once in LDS and shared by all 4 waves.
__global__ __launch_bounds__(128) void k_attn(const float* __restrict__ qkv,
                                              float* __restrict__ O) {
    __shared__ __align__(16) __bf16 Klds[32][64];      // [kv][d]   row-major
    __shared__ __align__(16) __bf16 Vlds[64][32];      // [d][kv]   transposed
    __shared__ __align__(16) __bf16 Pl[4][16][32];     // per-wave P staging
    int tid = threadIdx.x;
    int lane = tid & 31, half = lane >> 4, r = lane & 15;
    int wv = tid >> 5;
    int qb_blk = blockIdx.x * 64, h = blockIdx.y, b = blockIdx.z;
    int qb = qb_blk + wv * 16;
    const size_t ld = 3 * DIM;
    const float* base = qkv + (size_t)b * SEQ * ld;

    // Q fragments (A layout), head_dim 64 = two K=32 steps
    const float* qp = base + (size_t)(qb + r) * ld + h * HEAD_DIM;
    v16bf qf0, qf1;
    for (int e = 0; e < 16; ++e) {
        qf0[e] = f2bf(qp[kA(e, half)]);
        qf1[e] = f2bf(qp[32 + kA(e, half)]);
    }

    v8f o[4]; o[0] = v8f{}; o[1] = v8f{}; o[2] = v8f{}; o[3] = v8f{};
    float mrow[8], lrow[8];
    for (int v = 0; v < 8; ++v) { mrow[v] = -1e30f; lrow[v] = 0.f; }
    const float scale = 0.125f;   // 1/sqrt(64)
    const int c0_end = qb_blk + 63;

    for (int c0 = 0; c0 <= c0_end; c0 += 32) {
        // ---- stage K (row-major) and V (transposed) chunk, all 128 threads
        for (int j = 0; j < 4; ++j) {
            int f = tid + 128 * j;                 // float4 idx: 16 per 64-wide row
            int row = f >> 4, coff = (f & 15) * 4;
            const float* kp = base + (size_t)(c0 + row) * ld + DIM + h * HEAD_DIM + coff;
            const float* vp = base + (size_t)(c0 + row) * ld + 2 * DIM + h * HEAD_DIM + coff;
            float4 k4 = *(const float4*)kp;
            float4 v4 = *(const float4*)vp;
            uint2 pk; pk.x = pack2(k4.x, k4.y); pk.y = pack2(k4.z, k4.w);
            *(uint2*)&Klds[row][coff] = pk;
            Vlds[coff + 0][row] = f2bf(v4.x);
            Vlds[coff + 1][row] = f2bf(v4.y);
            Vlds[coff + 2][row] = f2bf(v4.z);
            Vlds[coff + 3][row] = f2bf(v4.w);
        }
        // prefetch next chunk's K/V rows into cache (global_prefetch_b8)
        if (c0 + 32 <= c0_end && tid < 64) {
            int row = c0 + 32 + (tid & 31);
            size_t off = (tid < 32) ? (size_t)DIM : (size_t)(2 * DIM);
            __builtin_prefetch(base + (size_t)row * ld + off + h * HEAD_DIM, 0, 1);
            __builtin_prefetch(base + (size_t)row * ld + off + h * HEAD_DIM + 32, 0, 1);
        }
        __syncthreads();

        if (c0 <= qb + 15) {                       // causal: this wave needs the chunk
            // ---- scores: two 16x16 tiles over kv columns [c0, c0+32)
            v8f s0 = {}, s1 = {};
            for (int sub = 0; sub < 2; ++sub) {
                v16bf kb0, kb1;
                for (int e = 0; e < 16; ++e) {
                    int d = kA(e, half);
                    kb0[e] = Klds[sub * 16 + r][d];
                    kb1[e] = Klds[sub * 16 + r][32 + d];
                }
                v8f s = {};
                s = wmma_bf16(qf0, kb0, s);
                s = wmma_bf16(qf1, kb1, s);
                if (sub == 0) s0 = s; else s1 = s;
            }
            // ---- online softmax (each C-layout row lives in one 16-lane half)
            int nc0 = c0 + r, nc1 = c0 + 16 + r;
            float p0[8], p1[8];
            for (int v = 0; v < 8; ++v) {
                int qrow = qb + v + 8 * half;
                float a0 = s0[v] * scale, a1 = s1[v] * scale;
                if (nc0 > qrow) a0 = -1e30f;
                if (nc1 > qrow) a1 = -1e30f;
                float mx = fmaxf(a0, a1);
                for (int m = 1; m < 16; m <<= 1) mx = fmaxf(mx, __shfl_xor(mx, m));
                float newm = fmaxf(mrow[v], mx);
                float f    = __expf(mrow[v] - newm);
                float e0   = __expf(a0 - newm);
                float e1   = __expf(a1 - newm);
                float rs   = e0 + e1;
                for (int m = 1; m < 16; m <<= 1) rs += __shfl_xor(rs, m);
                lrow[v] = lrow[v] * f + rs;
                mrow[v] = newm;
                o[0][v] *= f; o[1][v] *= f; o[2][v] *= f; o[3][v] *= f;
                p0[v] = e0; p1[v] = e1;
            }
            // ---- C-layout -> A-layout transpose of P via per-wave LDS
            for (int v = 0; v < 8; ++v) {
                Pl[wv][v + 8 * half][r]      = f2bf(p0[v]);
                Pl[wv][v + 8 * half][16 + r] = f2bf(p1[v]);
            }
            asm volatile("s_wait_dscnt 0x0" ::: "memory");
            v16bf pf;
            for (int e = 0; e < 16; ++e) pf[e] = Pl[wv][r][kA(e, half)];
            // ---- P @ V: K=32 kv rows, N=64 head dims (4 tiles)
            for (int nt = 0; nt < 4; ++nt) {
                v16bf vf;
                for (int e = 0; e < 16; ++e)
                    vf[e] = Vlds[nt * 16 + r][kA(e, half)];
                o[nt] = wmma_bf16(pf, vf, o[nt]);
            }
        }
        __syncthreads();
    }
    float* op = O + (size_t)b * SEQ * DIM;
    for (int v = 0; v < 8; ++v) {
        float inv = 1.0f / lrow[v];
        int qrow = qb + v + 8 * half;
        for (int nt = 0; nt < 4; ++nt)
            op[(size_t)qrow * DIM + h * HEAD_DIM + nt * 16 + r] = o[nt][v] * inv;
    }
}

// ---------------------------------------------------------------- driver
extern "C" void kernel_launch(void* const* d_in, const int* in_sizes, int n_in,
                              void* d_out, int out_size, void* d_ws, size_t ws_size,
                              hipStream_t stream) {
    const int*   ids   = (const int*)  d_in[0];
    const float* emb   = (const float*)d_in[1];
    const float* pos   = (const float*)d_in[2];
    const float* ln1_s = (const float*)d_in[3];
    const float* ln1_b = (const float*)d_in[4];
    const float* qkv_w = (const float*)d_in[5];
    const float* qkv_b = (const float*)d_in[6];
    const float* out_w = (const float*)d_in[7];
    const float* out_b = (const float*)d_in[8];
    const float* ln2_s = (const float*)d_in[9];
    const float* ln2_b = (const float*)d_in[10];
    const float* fc1_w = (const float*)d_in[11];
    const float* fc1_b = (const float*)d_in[12];
    const float* fc2_w = (const float*)d_in[13];
    const float* fc2_b = (const float*)d_in[14];
    const float* lnf_s = (const float*)d_in[15];
    const float* lnf_b = (const float*)d_in[16];
    float* logits = (float*)d_out;

    float* X    = (float*)d_ws;                 // [NTOK, DIM] residual stream
    float* H    = X    + (size_t)NTOK * DIM;    // [NTOK, DIM] ln scratch
    float* Obuf = H    + (size_t)NTOK * DIM;    // [NTOK, DIM] attention out
    float* QKV  = Obuf + (size_t)NTOK * DIM;    // [NTOK, 3*DIM]
    float* F    = QKV  + (size_t)NTOK * 3 * DIM;// [NTOK, DFF]

    dim3 blk256(256), blk128(128);

    k_embed<<<dim3(NTOK), blk256, 0, stream>>>(ids, emb, pos, X);

    for (int l = 0; l < LAYERS; ++l) {
        const float* qw  = qkv_w + (size_t)l * 3 * DIM * DIM;
        const float* qb  = qkv_b + (size_t)l * 3 * DIM;
        const float* ow  = out_w + (size_t)l * DIM * DIM;
        const float* ob  = out_b + (size_t)l * DIM;
        const float* f1w = fc1_w + (size_t)l * DFF * DIM;
        const float* f1b = fc1_b + (size_t)l * DFF;
        const float* f2w = fc2_w + (size_t)l * DIM * DFF;
        const float* f2b = fc2_b + (size_t)l * DIM;

        k_ln<<<dim3(NTOK), blk256, 0, stream>>>(X, ln1_s + l * DIM, ln1_b + l * DIM, H);
        k_gemm<<<dim3(3 * DIM / 64, NTOK / 128), blk256, 0, stream>>>(
            H, qw, qb, nullptr, QKV, NTOK, 3 * DIM, DIM, 0);
        k_attn<<<dim3(SEQ / 64, HEADS, BATCH), blk128, 0, stream>>>(QKV, Obuf);
        k_gemm<<<dim3(DIM / 64, NTOK / 128), blk256, 0, stream>>>(
            Obuf, ow, ob, X, X, NTOK, DIM, DIM, 0);          // residual add
        k_ln<<<dim3(NTOK), blk256, 0, stream>>>(X, ln2_s + l * DIM, ln2_b + l * DIM, H);
        k_gemm<<<dim3(DFF / 64, NTOK / 128), blk256, 0, stream>>>(
            H, f1w, f1b, nullptr, F, NTOK, DFF, DIM, 1);     // GELU
        k_gemm<<<dim3(DIM / 64, NTOK / 128), blk256, 0, stream>>>(
            F, f2w, f2b, X, X, NTOK, DIM, DFF, 0);           // residual add
    }

    k_ln<<<dim3(NTOK), blk256, 0, stream>>>(X, lnf_s, lnf_b, H);
    k_gemm<<<dim3(VOCAB / 64, NTOK / 128), blk256, 0, stream>>>(
        H, emb, nullptr, nullptr, logits, NTOK, VOCAB, DIM, 0);
}
